// MambaEncoder_76338748719400
// MI455X (gfx1250) — matrix operand order repaired and checked by reading, not compiled, not run on hardware
//
#include <hip/hip_runtime.h>

typedef _Float16 f16;
typedef __attribute__((ext_vector_type(16))) _Float16 v16h;
typedef __attribute__((ext_vector_type(8)))  _Float16 v8h;
typedef __attribute__((ext_vector_type(8)))  float    v8f;

#define DMODEL 1024
#define DINNER 2048
#define DTRANK 64
#define DSTATE 16
#define NLAYERS 3
#define NTOK    2048   // BATCH(2) * SEQ(1024)

__device__ __forceinline__ float sigmoidf_(float x) { return 1.0f / (1.0f + __expf(-x)); }

// ---------------- embedding gather ----------------
__global__ void k_embed(const int* __restrict__ ids, const float* __restrict__ emb,
                        float* __restrict__ x) {
    int tid = blockIdx.x * 256 + threadIdx.x;        // NTOK*DMODEL
    int tok = tid >> 10, d = tid & 1023;
    x[tid] = emb[ids[tok] * DMODEL + d];
}

// ---------------- RMSNorm (f16 out for GEMM input) ----------------
__global__ void k_rmsnorm_f16(const float* __restrict__ x, const float* __restrict__ w,
                              f16* __restrict__ out) {
    __shared__ float red[256];
    int tok = blockIdx.x, t = threadIdx.x;
    const float* row = x + (size_t)tok * DMODEL;
    float s = 0.f;
    for (int i = t; i < DMODEL; i += 256) { float v = row[i]; s += v * v; }
    red[t] = s; __syncthreads();
    for (int o = 128; o > 0; o >>= 1) { if (t < o) red[t] += red[t + o]; __syncthreads(); }
    float scale = rsqrtf(red[0] * (1.0f / DMODEL) + 1e-6f);
    for (int i = t; i < DMODEL; i += 256)
        out[(size_t)tok * DMODEL + i] = (f16)(row[i] * scale * w[i]);
}

// ---------------- RMSNorm (f32 out, final) ----------------
__global__ void k_rmsnorm_f32(const float* __restrict__ x, const float* __restrict__ w,
                              float* __restrict__ out) {
    __shared__ float red[256];
    int tok = blockIdx.x, t = threadIdx.x;
    const float* row = x + (size_t)tok * DMODEL;
    float s = 0.f;
    for (int i = t; i < DMODEL; i += 256) { float v = row[i]; s += v * v; }
    red[t] = s; __syncthreads();
    for (int o = 128; o > 0; o >>= 1) { if (t < o) red[t] += red[t + o]; __syncthreads(); }
    float scale = rsqrtf(red[0] * (1.0f / DMODEL) + 1e-6f);
    for (int i = t; i < DMODEL; i += 256)
        out[(size_t)tok * DMODEL + i] = row[i] * scale * w[i];
}

// ---------------- fp32 (K x N) -> f16 transposed (Npad x K), zero-padded ----------------
__global__ void k_transpose_f16(const float* __restrict__ W, f16* __restrict__ Wt,
                                int K, int N, int Npad) {
    int tid = blockIdx.x * 256 + threadIdx.x;        // Npad*K
    int n = tid / K, k = tid - n * K;
    Wt[tid] = (n < N) ? (f16)W[(size_t)k * N + n] : (f16)0.0f;
}

// ---------------- WMMA GEMM: C(MxN,f32) = A(MxK,f16 row-major) @ Bt(NxK,f16)^T ----------
// block = 256 threads = 8 waves; wave tile 16x64; block tile 32x256.
// M % 32 == 0, N % 64 == 0, K % 32 == 0 guaranteed by padding.
__global__ __launch_bounds__(256) void k_gemm_f16(
        const f16* __restrict__ A, const f16* __restrict__ Bt,
        float* __restrict__ C, int M, int N, int K) {
    int lane = threadIdx.x & 31;
    int wave = threadIdx.x >> 5;
    int wm = wave >> 2, wn = wave & 3;
    int m0 = blockIdx.y * 32 + wm * 16;
    int n0 = blockIdx.x * 256 + wn * 64;
    if (n0 >= N) return;                              // whole-wave guard only
    int l16 = lane & 15;
    int hi  = lane >> 4;                              // 0 | 1
    int ko  = hi ? 16 : 0;                            // A half-row K offset
    int kb  = hi ? 16 : 0;                            // B K offset
    const f16* arow = A + (size_t)(m0 + l16) * K;
    const f16* b0p = Bt + (size_t)(n0 +  0 + l16) * K + kb;
    const f16* b1p = Bt + (size_t)(n0 + 16 + l16) * K + kb;
    const f16* b2p = Bt + (size_t)(n0 + 32 + l16) * K + kb;
    const f16* b3p = Bt + (size_t)(n0 + 48 + l16) * K + kb;
    v8f acc0 = {}, acc1 = {}, acc2 = {}, acc3 = {};
    for (int k = 0; k < K; k += 32) {
        union { v16h v; v8h h[2]; } a;
        a.h[0] = *(const v8h*)(arow + k + ko);
        a.h[1] = *(const v8h*)(arow + k + ko + 8);
        v16h b0 = *(const v16h*)(b0p + k);
        v16h b1 = *(const v16h*)(b1p + k);
        v16h b2 = *(const v16h*)(b2p + k);
        v16h b3 = *(const v16h*)(b3p + k);
        acc0 = __builtin_amdgcn_wmma_f32_16x16x32_f16(false, a.v, false, b0, (short)0, acc0, false, false);
        acc1 = __builtin_amdgcn_wmma_f32_16x16x32_f16(false, a.v, false, b1, (short)0, acc1, false, false);
        acc2 = __builtin_amdgcn_wmma_f32_16x16x32_f16(false, a.v, false, b2, (short)0, acc2, false, false);
        acc3 = __builtin_amdgcn_wmma_f32_16x16x32_f16(false, a.v, false, b3, (short)0, acc3, false, false);
    }
    int rbase = m0 + (hi ? 8 : 0);                    // C layout: VGPR i -> row i (+8 for hi half)
#pragma unroll
    for (int i = 0; i < 8; i++) {
        size_t r = (size_t)(rbase + i) * N + n0 + l16;
        C[r +  0] = acc0[i];
        C[r + 16] = acc1[i];
        C[r + 32] = acc2[i];
        C[r + 48] = acc3[i];
    }
}

// ---------------- causal depthwise conv1d + SiLU -> u(f16) ----------------
__global__ void k_conv_silu(const float* __restrict__ xr, const float* __restrict__ cw,
                            const float* __restrict__ cb, f16* __restrict__ u) {
    int tid = blockIdx.x * 256 + threadIdx.x;         // NTOK*DINNER
    int tok = tid >> 11, e = tid & 2047;
    int b = tok >> 10, l = tok & 1023;
    float acc = cb[e];
#pragma unroll
    for (int j = 0; j < 4; j++) {
        int ll = l - 3 + j;
        if (ll >= 0) acc += cw[e * 4 + j] * xr[(size_t)(b * 1024 + ll) * 4096 + e];
    }
    u[tid] = (f16)(acc * sigmoidf_(acc));
}

// ---------------- dbc[:, 0:64] fp32 -> f16 compact ----------------
__global__ void k_dtlow(const float* __restrict__ dbc, f16* __restrict__ dtlow) {
    int tid = blockIdx.x * 256 + threadIdx.x;         // NTOK*64
    int r = tid >> 6, c = tid & 63;
    dtlow[tid] = (f16)dbc[(size_t)r * 128 + c];
}

// ---------------- softplus(dt + bias) in place ----------------
__global__ void k_softplus(float* __restrict__ dt, const float* __restrict__ bias) {
    int tid = blockIdx.x * 256 + threadIdx.x;         // NTOK*DINNER
    int e = tid & 2047;
    float v = dt[tid] + bias[e];
    dt[tid] = (v > 20.f) ? v : log1pf(__expf(v));
}

// ---------------- selective scan: lane = (channel, state); 16-lane reduce ------------
__global__ void k_scan(const f16* __restrict__ u, const float* __restrict__ dt,
                       const float* __restrict__ dbc, const float* __restrict__ Alog,
                       const float* __restrict__ Dv, float* __restrict__ y) {
    int tid = blockIdx.x * 256 + threadIdx.x;         // 2*2048*16 threads
    int n  = tid & 15;
    int ch = tid >> 4;                                // 0..4095
    int b = ch >> 11, e = ch & 2047;
    float A   = -__expf(Alog[e * 16 + n]);
    float Dval = Dv[e];
    float h = 0.f;
    size_t tokbase = (size_t)b * 1024;
    for (int t = 0; t < 1024; t++) {
        size_t row = tokbase + t;
        size_t idx = row * 2048 + e;
        float dtv = dt[idx];
        float uv  = (float)u[idx];
        float Bv  = dbc[row * 128 + 64 + n];
        float Cv  = dbc[row * 128 + 80 + n];
        h = h * __expf(dtv * A) + dtv * uv * Bv;
        float p = h * Cv;
        p += __shfl_xor(p, 8, 32);
        p += __shfl_xor(p, 4, 32);
        p += __shfl_xor(p, 2, 32);
        p += __shfl_xor(p, 1, 32);
        if (n == 0) y[idx] = p + uv * Dval;
    }
}

// ---------------- gate: yg = f16(y * silu(res)) ----------------
__global__ void k_gate(const float* __restrict__ y, const float* __restrict__ xr,
                       f16* __restrict__ yg) {
    int tid = blockIdx.x * 256 + threadIdx.x;         // NTOK*DINNER
    int tok = tid >> 11, e = tid & 2047;
    float r = xr[(size_t)tok * 4096 + 2048 + e];
    yg[tid] = (f16)(y[tid] * (r * sigmoidf_(r)));
}

// ---------------- residual add ----------------
__global__ void k_add(float* __restrict__ x, const float* __restrict__ t) {
    int tid = blockIdx.x * 256 + threadIdx.x;         // NTOK*DMODEL
    x[tid] += t[tid];
}

extern "C" void kernel_launch(void* const* d_in, const int* in_sizes, int n_in,
                              void* d_out, int out_size, void* d_ws, size_t ws_size,
                              hipStream_t stream) {
    (void)in_sizes; (void)n_in; (void)out_size; (void)ws_size;
    const int*   ids  = (const int*)  d_in[0];
    const float* emb  = (const float*)d_in[1];
    const float* ipw  = (const float*)d_in[2];   // 3 x 1024 x 4096
    const float* cw   = (const float*)d_in[3];   // 3 x 2048 x 1 x 4
    const float* cb   = (const float*)d_in[4];   // 3 x 2048
    const float* xpw  = (const float*)d_in[5];   // 3 x 2048 x 96
    const float* dpw  = (const float*)d_in[6];   // 3 x 64 x 2048
    const float* dpb  = (const float*)d_in[7];   // 3 x 2048
    const float* alog = (const float*)d_in[8];   // 3 x 2048 x 16
    const float* Dp   = (const float*)d_in[9];   // 3 x 2048
    const float* opw  = (const float*)d_in[10];  // 3 x 2048 x 1024
    const float* nw   = (const float*)d_in[11];  // 1024
    float* out        = (float*)d_out;

    char* ws = (char*)d_ws;
    f16*   wT    = (f16*)  (ws);                  //  8 MB (reused per-GEMM)
    float* x     = (float*)(ws +   8388608);      //  8 MB residual
    f16*   xn    = (f16*)  (ws +  16777216);      //  4 MB normed f16
    float* xr    = (float*)(ws +  20971520);      // 32 MB in_proj out (xb|res)
    f16*   u     = (f16*)  (ws +  54525952);      //  8 MB
    float* dbc   = (float*)(ws +  62914560);      //  1 MB (N padded to 128)
    f16*   dtlow = (f16*)  (ws +  63963136);      //  0.25 MB
    float* dt    = (float*)(ws +  64225280);      // 16 MB
    float* y     = (float*)(ws +  81002496);      // 16 MB
    f16*   yg    = (f16*)  (ws +  97779712);      //  8 MB
    float* tmp   = (float*)(ws + 106168320);      //  8 MB

    k_embed<<<NTOK * DMODEL / 256, 256, 0, stream>>>(ids, emb, x);

    for (int l = 0; l < NLAYERS; l++) {
        // RMSNorm -> f16
        k_rmsnorm_f16<<<NTOK, 256, 0, stream>>>(x, nw, xn);

        // in_proj: (2048x1024) @ (1024x4096)
        k_transpose_f16<<<4096 * 1024 / 256, 256, 0, stream>>>(
            ipw + (size_t)l * 1024 * 4096, wT, 1024, 4096, 4096);
        k_gemm_f16<<<dim3(16, 64), 256, 0, stream>>>(xn, wT, xr, 2048, 4096, 1024);

        // causal depthwise conv + silu -> u(f16)
        k_conv_silu<<<NTOK * DINNER / 256, 256, 0, stream>>>(
            xr, cw + (size_t)l * DINNER * 4, cb + (size_t)l * DINNER, u);

        // x_proj: (2048x2048) @ (2048x96) padded to N=128
        k_transpose_f16<<<128 * 2048 / 256, 256, 0, stream>>>(
            xpw + (size_t)l * 2048 * 96, wT, 2048, 96, 128);
        k_gemm_f16<<<dim3(1, 64), 256, 0, stream>>>(u, wT, dbc, 2048, 128, 2048);

        // dt_proj: (2048x64) @ (64x2048), then softplus(+bias)
        k_dtlow<<<NTOK * 64 / 256, 256, 0, stream>>>(dbc, dtlow);
        k_transpose_f16<<<2048 * 64 / 256, 256, 0, stream>>>(
            dpw + (size_t)l * 64 * 2048, wT, 64, 2048, 2048);
        k_gemm_f16<<<dim3(8, 64), 256, 0, stream>>>(dtlow, wT, dt, 2048, 2048, 64);
        k_softplus<<<NTOK * DINNER / 256, 256, 0, stream>>>(dt, dpb + (size_t)l * DINNER);

        // selective scan
        k_scan<<<256, 256, 0, stream>>>(u, dt, dbc,
            alog + (size_t)l * DINNER * DSTATE, Dp + (size_t)l * DINNER, y);

        // gate + out_proj + residual
        k_gate<<<NTOK * DINNER / 256, 256, 0, stream>>>(y, xr, yg);
        k_transpose_f16<<<1024 * 2048 / 256, 256, 0, stream>>>(
            opw + (size_t)l * 2048 * 1024, wT, 2048, 1024, 1024);
        k_gemm_f16<<<dim3(4, 64), 256, 0, stream>>>(yg, wT, tmp, 2048, 1024, 2048);
        k_add<<<NTOK * DMODEL / 256, 256, 0, stream>>>(x, tmp);
    }

    k_rmsnorm_f32<<<NTOK, 256, 0, stream>>>(x, nw, out);
}
